// MultiHeadedAttentionSANM_90623809945918
// MI455X (gfx1250) — matrix operand order repaired and checked by reading, not compiled, not run on hardware
//
#include <hip/hip_runtime.h>

typedef __attribute__((ext_vector_type(16))) _Float16 v16h;
typedef __attribute__((ext_vector_type(8)))  _Float16 v8h;
typedef __attribute__((ext_vector_type(8)))  float    v8f;

#define NFEAT   512
#define TQKV    1536
#define BB      16
#define TT      1024
#define MTOT    (BB * TT)      // 16384
#define DKH     128
#define NEGV    (-1e9f)

// A-fragment slot map (16-bit A 16x32, ISA 7.12.2):
//   this code uses K = lh*16 + s for slots 0-15 (contiguous, not the ISA table order)
// B-fragment slot map (16-bit B 32x16): lane-half lh holds K = lh*16 + s, same order as A (this code's choice).

// ---------------------------------------------------------------------------
// Kernel 1: QKV = X @ Wqkv + b. Q,K,V row-major f16; V additionally written
// transposed (vT[b][f][t]) with packed 16B stores so attention's P@V
// B-fragments are contiguous. Epilogue routing by block-uniform col0.
// ---------------------------------------------------------------------------
__global__ __launch_bounds__(128)
void qkv_gemm_kernel(const float* __restrict__ x, const float* __restrict__ w,
                     const float* __restrict__ bias,
                     _Float16* __restrict__ qf, _Float16* __restrict__ kf,
                     _Float16* __restrict__ vf, _Float16* __restrict__ vT) {
    __shared__ v16h fA[64][2];   // [row][lh]
    __shared__ v16h fB[64][2];   // [col][lh]
    const int tid  = threadIdx.x;
    const int lane = tid & 31;
    const int wave = tid >> 5;
    const int lh   = lane >> 4;
    const int ln   = lane & 15;
    const int row0 = blockIdx.x * 64;
    const int col0 = blockIdx.y * 64;

    const int ar  = tid >> 1;    // A staging: row 0..63
    const int ach = tid & 1;     //            col-half: c = ach*16 + i
    const int bc  = tid & 63;    // B staging: col 0..63
    const int bkh = tid >> 6;    //            k-half: k = bkh*16 + i

    v8f acc[4];
#pragma unroll
    for (int i = 0; i < 4; ++i) { v8f z = {}; acc[i] = z; }

    for (int kk = 0; kk < NFEAT; kk += 32) {
        {   // stage A: x[row0+ar][kk + ach*16 + 0..15]  (contiguous 64B global)
            const float* src = x + (size_t)(row0 + ar) * NFEAT + kk + ach * 16;
            v8h lo, hi;
#pragma unroll
            for (int i = 0; i < 8; ++i) {
                lo[i] = (_Float16)src[i];
                hi[i] = (_Float16)src[8 + i];
            }
            *(v8h*)(((_Float16*)&fA[ar][ach]) + 0) = lo;
            *(v8h*)(((_Float16*)&fA[ar][ach]) + 8) = hi;
        }
        {   // stage B: w[kk + bkh*16 + i][col0 + bc]  (lane-coalesced global)
            v16h t;
#pragma unroll
            for (int i = 0; i < 16; ++i)
                t[i] = (_Float16)w[(size_t)(kk + bkh * 16 + i) * TQKV + col0 + bc];
            fB[bc][bkh] = t;
        }
        __syncthreads();
        v16h af = fA[wave * 16 + ln][lh];
#pragma unroll
        for (int nt = 0; nt < 4; ++nt) {
            v16h bf = fB[nt * 16 + ln][lh];
            acc[nt] = __builtin_amdgcn_wmma_f32_16x16x32_f16(
                false, af, false, bf, (short)0, acc[nt], false, false);
        }
        __syncthreads();
    }

    const int mbase = row0 + wave * 16 + lh * 8;
    if (col0 < 512) {            // ---- Q tile (uniform branch) ----
#pragma unroll
        for (int nt = 0; nt < 4; ++nt) {
            int ncol = col0 + nt * 16 + ln;
            float bv = bias[ncol];
#pragma unroll
            for (int r = 0; r < 8; ++r)
                qf[(size_t)(mbase + r) * NFEAT + ncol] = (_Float16)(acc[nt][r] + bv);
        }
    } else if (col0 < 1024) {    // ---- K tile ----
#pragma unroll
        for (int nt = 0; nt < 4; ++nt) {
            int ncol = col0 + nt * 16 + ln;
            float bv = bias[ncol];
            int f = ncol - 512;
#pragma unroll
            for (int r = 0; r < 8; ++r)
                kf[(size_t)(mbase + r) * NFEAT + f] = (_Float16)(acc[nt][r] + bv);
        }
    } else {                     // ---- V tile: row-major + packed transposed copy ----
        const int b = mbase >> 10;
        const int t = mbase & 1023;       // multiple of 8 -> 16B-aligned vT run
#pragma unroll
        for (int nt = 0; nt < 4; ++nt) {
            int ncol = col0 + nt * 16 + ln;
            float bv = bias[ncol];
            int f = ncol - 1024;
            v8h hv8;
#pragma unroll
            for (int r = 0; r < 8; ++r) {
                _Float16 hv = (_Float16)(acc[nt][r] + bv);
                hv8[r] = hv;
                vf[(size_t)(mbase + r) * NFEAT + f] = hv;
            }
            *(v8h*)(vT + ((size_t)b * NFEAT + f) * TT + t) = hv8;   // vT[b][f][t..t+7]
        }
    }
}

// ---------------------------------------------------------------------------
// Kernel 2: FSMN depthwise conv (k=11, LEFT_PAD=5) with mask + residual.
// ---------------------------------------------------------------------------
__global__ __launch_bounds__(512)
void fsmn_kernel(const _Float16* __restrict__ vf, const float* __restrict__ mask,
                 const float* __restrict__ wf, float* __restrict__ out) {
    const int m = blockIdx.x;
    const int f = threadIdx.x;
    const int b = m >> 10;
    const int t = m & 1023;
    const _Float16* vcol = vf + f;
    const float*    mrow = mask + (size_t)b * TT;

    float y = (float)vcol[(size_t)m * NFEAT] * mrow[t];   // residual
#pragma unroll
    for (int j = 0; j < 11; ++j) {
        int tt = t + j - 5;
        if (tt >= 0 && tt < TT) {
            float xin = (float)vcol[(size_t)(b * TT + tt) * NFEAT] * mrow[tt];
            y += wf[f * 11 + j] * xin;
        }
    }
    out[(size_t)m * NFEAT + f] = y * mrow[t];
}

// ---------------------------------------------------------------------------
// Kernel 3: masked flash attention. One wave per (b, h, 16-row q tile).
// 32 keys/step: 8 WMMAs scores + online softmax + 8 WMMAs ctx.
// All operand loads and staging copies are wide (v16h / v8h).
// ---------------------------------------------------------------------------
__global__ __launch_bounds__(32)
void attn_kernel(const _Float16* __restrict__ qf, const _Float16* __restrict__ kf,
                 const _Float16* __restrict__ vT, const float* __restrict__ mask,
                 _Float16* __restrict__ ctx) {
    __shared__ v16h  fQ[4][16][2];   // [dc][row][lh]   A-fragments of Q
    __shared__ v16h  fK[32][8];      // [token][d/16]   row-major K
    __shared__ v16h  fV[8][16][2];   // [dc][n][lh]     B-fragments of V
    __shared__ float ldsS[16][32];
    __shared__ v16h  fP[16][2];      // [row][lh]       A-fragments of P
    __shared__ float ldsAl[16];
    __shared__ float ldsL[16];

    const int lane = threadIdx.x;
    const int lh   = lane >> 4;
    const int ln   = lane & 15;
    const int qt   = blockIdx.x;
    const int h    = blockIdx.y;
    const int b    = blockIdx.z;
    const size_t base  = (size_t)b * TT * NFEAT + (size_t)h * DKH;
    const _Float16* vtb = vT + ((size_t)b * NFEAT + (size_t)h * DKH) * TT;
    const float scale = 0.08838834764831845f;   // 1/sqrt(128)

    {   // stage Q tile 16x128: each global v8h chunk is one 16B fragment slot
        const _Float16* src = qf + base + (size_t)(qt * 16 + ln) * NFEAT + lh * 64;
#pragma unroll
        for (int j = 0; j < 8; ++j) {
            v8h g = *(const v8h*)(src + j * 8);
            int c   = lh * 64 + j * 8;
            int dc  = c >> 5;
            int cc  = c & 31;
            int lhd = cc >> 4;
            int so  = cc & 8;
            *(v8h*)(((_Float16*)&fQ[dc][ln][lhd]) + so) = g;
        }
    }
    __syncthreads();
    v16h qfr[4];
#pragma unroll
    for (int dc = 0; dc < 4; ++dc) qfr[dc] = fQ[dc][ln][lh];

    v8f acc[8];
#pragma unroll
    for (int i = 0; i < 8; ++i) { v8f z = {}; acc[i] = z; }
    float mrun = -3.0e38f;
    float lrun = 0.0f;

    for (int ks = 0; ks < TT / 32; ++ks) {
        const int kt0 = ks * 32;
        __syncthreads();
        {   // stage K row-major: lane -> token kt0+lane, 8 x v16h wide copies
            const v16h* ksrc = (const v16h*)(kf + base + (size_t)(kt0 + lane) * NFEAT);
#pragma unroll
            for (int j = 0; j < 8; ++j) fK[lane][j] = ksrc[j];
        }
        {   // stage V fragments from transposed vT: 8 wide copies per lane
            // fV[d>>4][d&15][lhv] slot s = V[k=lhv*16+s][d] = vT[d][kt0+lhv*16+s]
#pragma unroll
            for (int i = 0; i < 8; ++i) {
                int e   = lane * 8 + i;       // 0..255
                int d   = e >> 1;
                int lhv = e & 1;
                fV[d >> 4][d & 15][lhv] =
                    *(const v16h*)(vtb + (size_t)d * TT + kt0 + lhv * 16);
            }
        }
        __syncthreads();
        // scores: two 16-key column tiles, K-dim = 128 in 4 WMMA steps each
#pragma unroll
        for (int ct = 0; ct < 2; ++ct) {
            v8f s8 = {};
#pragma unroll
            for (int dc = 0; dc < 4; ++dc) {
                v16h bf = fK[ct * 16 + ln][dc * 2 + lh];   // d = dc*32 + lh*16 + s
                s8 = __builtin_amdgcn_wmma_f32_16x16x32_f16(
                    false, qfr[dc], false, bf, (short)0, s8, false, false);
            }
            const int tcol = kt0 + ct * 16 + ln;
            const float mv = mask[(size_t)b * TT + tcol];
#pragma unroll
            for (int r = 0; r < 8; ++r) {
                float v = s8[r] * scale;
                if (mv == 0.0f) v = NEGV;
                ldsS[lh * 8 + r][ct * 16 + ln] = v;
            }
        }
        __syncthreads();
        if (lane < 16) {   // online softmax: one q-row per lane
            float sv[32];
#pragma unroll
            for (int j = 0; j < 32; ++j) sv[j] = ldsS[lane][j];
            float mx = mrun;
#pragma unroll
            for (int j = 0; j < 32; ++j) mx = fmaxf(mx, sv[j]);
            float al = __expf(mrun - mx);
            float sum = 0.0f;
            v8h pv[4];   // chunk c: (lh = c>>1, off = (c&1)*8)
#pragma unroll
            for (int j = 0; j < 32; ++j) {
                float p = __expf(sv[j] - mx);
                sum += p;
                pv[j >> 3][j & 7] = (_Float16)p;
            }
            *(v8h*)(((_Float16*)&fP[lane][0]) + 0) = pv[0];
            *(v8h*)(((_Float16*)&fP[lane][0]) + 8) = pv[1];
            *(v8h*)(((_Float16*)&fP[lane][1]) + 0) = pv[2];
            *(v8h*)(((_Float16*)&fP[lane][1]) + 8) = pv[3];
            lrun = lrun * al + sum;
            mrun = mx;
            ldsAl[lane] = al;
        }
        __syncthreads();
        float al8[8];
#pragma unroll
        for (int r = 0; r < 8; ++r) al8[r] = ldsAl[lh * 8 + r];
#pragma unroll
        for (int dc = 0; dc < 8; ++dc)
#pragma unroll
            for (int r = 0; r < 8; ++r) acc[dc][r] *= al8[r];
        v16h pf = fP[ln][lh];
#pragma unroll
        for (int dc = 0; dc < 8; ++dc) {
            v16h bv = fV[dc][ln][lh];
            acc[dc] = __builtin_amdgcn_wmma_f32_16x16x32_f16(
                false, pf, false, bv, (short)0, acc[dc], false, false);
        }
    }
    if (lane < 16) ldsL[lane] = lrun;
    __syncthreads();
#pragma unroll
    for (int dc = 0; dc < 8; ++dc) {
#pragma unroll
        for (int r = 0; r < 8; ++r) {
            const int m = qt * 16 + lh * 8 + r;
            float val = acc[dc][r] / ldsL[lh * 8 + r];
            ctx[base + (size_t)m * NFEAT + dc * 16 + ln] = (_Float16)val;
        }
    }
}

// ---------------------------------------------------------------------------
// Kernel 4: out = ctx @ Wout + b_out + fsmn_mem   (f32 output)
// ---------------------------------------------------------------------------
__global__ __launch_bounds__(128)
void out_gemm_kernel(const _Float16* __restrict__ ctx, const float* __restrict__ w,
                     const float* __restrict__ bias, const float* __restrict__ fsmn,
                     float* __restrict__ out) {
    __shared__ v16h fA[64][2];
    __shared__ v16h fB[64][2];
    const int tid  = threadIdx.x;
    const int lane = tid & 31;
    const int wave = tid >> 5;
    const int lh   = lane >> 4;
    const int ln   = lane & 15;
    const int row0 = blockIdx.x * 64;
    const int col0 = blockIdx.y * 64;

    const int ar  = tid >> 1;
    const int ach = tid & 1;
    const int bc  = tid & 63;
    const int bkh = tid >> 6;

    v8f acc[4];
#pragma unroll
    for (int i = 0; i < 4; ++i) { v8f z = {}; acc[i] = z; }

    for (int kk = 0; kk < NFEAT; kk += 32) {
        {   // stage A from f16 ctx: one 32B global load, two 16B LDS stores
            const _Float16* src = ctx + (size_t)(row0 + ar) * NFEAT + kk + ach * 16;
            v16h g = *(const v16h*)src;
            v8h lo = __builtin_shufflevector(g, g, 0, 1, 2, 3, 4, 5, 6, 7);
            v8h hi = __builtin_shufflevector(g, g, 8, 9, 10, 11, 12, 13, 14, 15);
            *(v8h*)(((_Float16*)&fA[ar][ach]) + 0) = lo;
            *(v8h*)(((_Float16*)&fA[ar][ach]) + 8) = hi;
        }
        {   // stage B
            v16h t;
#pragma unroll
            for (int i = 0; i < 16; ++i)
                t[i] = (_Float16)w[(size_t)(kk + bkh * 16 + i) * NFEAT + col0 + bc];
            fB[bc][bkh] = t;
        }
        __syncthreads();
        v16h af = fA[wave * 16 + ln][lh];
#pragma unroll
        for (int nt = 0; nt < 4; ++nt) {
            v16h bf = fB[nt * 16 + ln][lh];
            acc[nt] = __builtin_amdgcn_wmma_f32_16x16x32_f16(
                false, af, false, bf, (short)0, acc[nt], false, false);
        }
        __syncthreads();
    }
    const int mbase = row0 + wave * 16 + lh * 8;
#pragma unroll
    for (int nt = 0; nt < 4; ++nt) {
        int ncol = col0 + nt * 16 + ln;
        float bv = bias[ncol];
#pragma unroll
        for (int r = 0; r < 8; ++r) {
            size_t idx = (size_t)(mbase + r) * NFEAT + ncol;
            out[idx] = acc[nt][r] + bv + fsmn[idx];
        }
    }
}

// ---------------------------------------------------------------------------
extern "C" void kernel_launch(void* const* d_in, const int* in_sizes, int n_in,
                              void* d_out, int out_size, void* d_ws, size_t ws_size,
                              hipStream_t stream) {
    const float* x      = (const float*)d_in[0];
    const float* mask   = (const float*)d_in[1];
    const float* w_qkv  = (const float*)d_in[2];
    const float* b_qkv  = (const float*)d_in[3];
    const float* w_out  = (const float*)d_in[4];
    const float* b_out  = (const float*)d_in[5];
    const float* w_fsmn = (const float*)d_in[6];
    float* out = (float*)d_out;

    char* ws = (char*)d_ws;
    const size_t HBUF = (size_t)MTOT * NFEAT * sizeof(_Float16);  // 16 MB
    _Float16* qfb  = (_Float16*)(ws);
    _Float16* kfb  = (_Float16*)(ws + 1 * HBUF);
    _Float16* vfb  = (_Float16*)(ws + 2 * HBUF);
    _Float16* ctxb = (_Float16*)(ws + 3 * HBUF);
    _Float16* vTb  = (_Float16*)(ws + 4 * HBUF);                  // transposed V
    float*    fsmn = (float*)   (ws + 5 * HBUF);                  // 32 MB f32

    qkv_gemm_kernel<<<dim3(MTOT / 64, TQKV / 64), 128, 0, stream>>>(
        x, w_qkv, b_qkv, qfb, kfb, vfb, vTb);
    fsmn_kernel<<<dim3(MTOT), 512, 0, stream>>>(vfb, mask, w_fsmn, fsmn);
    attn_kernel<<<dim3(TT / 16, 4, BB), 32, 0, stream>>>(qfb, kfb, vTb, mask, ctxb);
    out_gemm_kernel<<<dim3(MTOT / 64, NFEAT / 64), 128, 0, stream>>>(
        ctxb, w_out, b_out, fsmn, out);
}
